// DecoderBlock_29540785062647
// MI455X (gfx1250) — matrix operand order repaired and checked by reading, not compiled, not run on hardware
//
#include <hip/hip_runtime.h>
#include <math.h>

typedef long long ll;

// ---------------- CDNA5 WMMA types ----------------
typedef __bf16 bf16_t;
typedef bf16_t v16bf __attribute__((ext_vector_type(16)));
typedef bf16_t v2bf  __attribute__((ext_vector_type(2)));
typedef float  v8f  __attribute__((ext_vector_type(8)));

#define LSTR 36   // 32 K-elems + 4 pad (bf16 units): keeps ds_store_b64 8B-aligned

// packed f32x2 -> bf16x2 (lowers to v_cvt_pk_bf16_f32)
__device__ __forceinline__ unsigned pack_bf2(float x, float y) {
  v2bf p; p.x = (bf16_t)x; p.y = (bf16_t)y;
  return __builtin_bit_cast(unsigned, p);
}
__device__ __forceinline__ unsigned short bf1(float x) {
  return __builtin_bit_cast(unsigned short, (bf16_t)x);
}

union FragBF { v16bf v; unsigned int u[8]; };

// A fragment 16x32 bf16: lane L -> row m = L%16, half = L/16.
// VGPR i: K = (i/4)*16 + half*8 + (i%4)*2 (+1 in high half). ISA 7.12.2.
__device__ __forceinline__ v16bf frag_a(const unsigned short* s, int r0, int lane) {
  FragBF f;
  int m = r0 + (lane & 15);
  int half = lane >> 4;
  #pragma unroll
  for (int i = 0; i < 8; ++i) {
    int klo = (i >> 2) * 16 + half * 8 + (i & 3) * 2;
    f.u[i] = *(const unsigned int*)&s[m * LSTR + klo];
  }
  return f.v;
}

// B fragment 32x16 bf16 (tile stored n-major: sB[n][k]): lane -> col n = L%16,
// VGPR i packs K = (i/4)*16 + half*8 + (i%4)*2 (+1), matching frag_a.
__device__ __forceinline__ v16bf frag_b(const unsigned short* s, int n0, int lane) {
  FragBF f;
  int n = n0 + (lane & 15);
  int half = lane >> 4;
  #pragma unroll
  for (int i = 0; i < 8; ++i) {
    int klo = (i >> 2) * 16 + half * 8 + (i & 3) * 2;
    f.u[i] = *(const unsigned int*)&s[n * LSTR + klo];
  }
  return f.v;
}

__device__ __forceinline__ v8f wmma_bf16(v16bf a, v16bf b, v8f c) {
  return __builtin_amdgcn_wmma_f32_16x16x32_bf16(
      /*neg_a=*/false, a, /*neg_b=*/false, b,
      /*c_mod=*/(short)0, c, /*reuse_a=*/false, /*reuse_b=*/false);
}

// ---------------- Generic batched strided bf16-WMMA GEMM ----------------
// C[z](m,n) = sum_k A[z](m,k) * B[z](k,n);  z -> (outer=z/idiv, inner=z%idiv)
// element(A) = A + outer*a_os + inner*a_is + m*a_rs + k*a_cs   (fp32)
// element(B) = B + outer*b_os + inner*b_is + k*b_rs + n*b_cs   (fp32)
// element(C) = C + outer*c_os + inner*c_is + m*c_rs + n        (fp32)
// Block tile 128x128, K-step 32, 256 threads (8 waves, 4x2, 32x64 per wave).
// AK: a_cs==1 (K-contiguous A); !AK requires a_rs==1 (M-contiguous A).
// BK: b_rs==1 (K-contiguous B); !BK requires b_cs==1 (N-contiguous B).
// Requires M%128==0, N%128==0, K%32==0.
template <bool AK, bool BK>
__global__ __launch_bounds__(256) void gemm_bf16_wmma(
    const float* __restrict__ A, const float* __restrict__ B, float* __restrict__ C,
    int M, int N, int K,
    ll a_rs, ll a_cs, ll a_os, ll a_is,
    ll b_rs, ll b_cs, ll b_os, ll b_is,
    ll c_rs, ll c_os, ll c_is,
    int idiv)
{
  __shared__ unsigned short sA[128 * LSTR];
  __shared__ unsigned short sB[128 * LSTR];

  int z = blockIdx.z;
  int zo = z / idiv, zi = z % idiv;
  A += (ll)zo * a_os + (ll)zi * a_is;
  B += (ll)zo * b_os + (ll)zi * b_is;
  C += (ll)zo * c_os + (ll)zi * c_is;

  const int m0 = blockIdx.y * 128;
  const int n0 = blockIdx.x * 128;
  const int tid  = threadIdx.x;
  const int wid  = tid >> 5;
  const int lane = tid & 31;
  const int wr = (wid >> 1) * 32;   // wave row offset within 128x128 tile
  const int wc = (wid & 1) * 64;    // wave col offset

  v8f zero = {0.f,0.f,0.f,0.f,0.f,0.f,0.f,0.f};
  v8f acc[2][4];
  #pragma unroll
  for (int i = 0; i < 2; ++i)
    #pragma unroll
    for (int j = 0; j < 4; ++j) acc[i][j] = zero;

  // uniform K-offsets kept incremental (stay in SGPRs; no per-iter 64-bit muls)
  ll aK = 0, bK = 0;
  const ll aKstep = 32 * a_cs;
  const ll bKstep = 32 * b_rs;

  for (int kk = 0; kk < K; kk += 32) {
    // ---- stage A tile 128x32 (row-major in LDS) ----
    if constexpr (AK) {                    // K-contiguous: float4 along K
      #pragma unroll
      for (int j = 0; j < 4; ++j) {
        int idx = tid + j * 256;           // 0..1023 float4s
        int rr = idx >> 3;                 // row 0..127
        int c4 = (idx & 7) * 4;            // col 0,4,..,28
        float4 v = *(const float4*)&A[aK + (ll)(m0 + rr) * a_rs + c4];
        uint2 p;
        p.x = pack_bf2(v.x, v.y);
        p.y = pack_bf2(v.z, v.w);
        *(uint2*)&sA[rr * LSTR + c4] = p;
      }
    } else {                               // M-contiguous (transposed view): float4 along M
      #pragma unroll
      for (int j = 0; j < 4; ++j) {
        int idx = tid + j * 256;
        int m4 = (idx & 31) * 4;           // rows m4..m4+3
        int cc = idx >> 5;                 // col 0..31
        float4 v = *(const float4*)&A[aK + (ll)cc * a_cs + (m0 + m4)];
        sA[(m4 + 0) * LSTR + cc] = bf1(v.x);
        sA[(m4 + 1) * LSTR + cc] = bf1(v.y);
        sA[(m4 + 2) * LSTR + cc] = bf1(v.z);
        sA[(m4 + 3) * LSTR + cc] = bf1(v.w);
      }
    }
    // ---- stage B tile as Bt[n][k] (n-major in LDS) ----
    if constexpr (BK) {                    // K-contiguous: float4 along K
      #pragma unroll
      for (int j = 0; j < 4; ++j) {
        int idx = tid + j * 256;
        int nn = idx >> 3;                 // n 0..127
        int k4 = (idx & 7) * 4;            // k 0,4,..,28
        float4 v = *(const float4*)&B[bK + (ll)(n0 + nn) * b_cs + k4];
        uint2 p;
        p.x = pack_bf2(v.x, v.y);
        p.y = pack_bf2(v.z, v.w);
        *(uint2*)&sB[nn * LSTR + k4] = p;
      }
    } else {                               // N-contiguous: float4 along N
      #pragma unroll
      for (int j = 0; j < 4; ++j) {
        int idx = tid + j * 256;
        int n4 = (idx & 31) * 4;           // n n4..n4+3
        int kkl = idx >> 5;                // k 0..31
        float4 v = *(const float4*)&B[bK + (ll)kkl * b_rs + (n0 + n4)];
        sB[(n4 + 0) * LSTR + kkl] = bf1(v.x);
        sB[(n4 + 1) * LSTR + kkl] = bf1(v.y);
        sB[(n4 + 2) * LSTR + kkl] = bf1(v.z);
        sB[(n4 + 3) * LSTR + kkl] = bf1(v.w);
      }
    }
    if (kk + 32 < K) {  // gfx1250: lowers to global_prefetch_b8
      __builtin_prefetch(A + aK + aKstep + (ll)(m0 + (tid >> 1)) * a_rs + (ll)((tid & 1) * 16) * a_cs, 0, 0);
      __builtin_prefetch(B + bK + bKstep + (ll)((tid & 31)) * b_rs + (ll)(n0 + (tid >> 5) * 16) * b_cs, 0, 0);
    }
    __syncthreads();

    v16bf a0 = frag_a(sA, wr +  0, lane);
    v16bf a1 = frag_a(sA, wr + 16, lane);
    v16bf b0 = frag_b(sB, wc +  0, lane);
    v16bf b1 = frag_b(sB, wc + 16, lane);
    v16bf b2 = frag_b(sB, wc + 32, lane);
    v16bf b3 = frag_b(sB, wc + 48, lane);
    acc[0][0] = wmma_bf16(a0, b0, acc[0][0]);
    acc[0][1] = wmma_bf16(a0, b1, acc[0][1]);
    acc[0][2] = wmma_bf16(a0, b2, acc[0][2]);
    acc[0][3] = wmma_bf16(a0, b3, acc[0][3]);
    acc[1][0] = wmma_bf16(a1, b0, acc[1][0]);
    acc[1][1] = wmma_bf16(a1, b1, acc[1][1]);
    acc[1][2] = wmma_bf16(a1, b2, acc[1][2]);
    acc[1][3] = wmma_bf16(a1, b3, acc[1][3]);
    __syncthreads();

    aK += aKstep;
    bK += bKstep;
  }

  // C/D 16x16 f32 layout: VGPR r, lanes 0-15 -> row r, lanes 16-31 -> row r+8
  const int half = lane >> 4;
  const int cidx = lane & 15;
  #pragma unroll
  for (int fi = 0; fi < 2; ++fi) {
    #pragma unroll
    for (int fj = 0; fj < 4; ++fj) {
      v8f a = acc[fi][fj];
      #pragma unroll
      for (int r = 0; r < 8; ++r) {
        int row = m0 + wr + fi * 16 + r + half * 8;
        int col = n0 + wc + fj * 16 + cidx;
        C[(ll)row * c_rs + col] = a[r];
      }
    }
  }
}

// ---------------- softmax over 128-wide rows (head-dim softmax) ----------------
// x <- softmax(x * inv [, -inf where col > row%maskS]) ; one wave32 per row.
__global__ __launch_bounds__(256) void softmax128_k(float* __restrict__ x, ll rows,
                                                    int maskS, float inv)
{
  int w = threadIdx.x >> 5, lane = threadIdx.x & 31;
  ll row = (ll)blockIdx.x * 8 + w;
  if (row >= rows) return;
  float* p = x + row * 128;
  float4 v = ((const float4*)p)[lane];
  float e[4] = {v.x * inv, v.y * inv, v.z * inv, v.w * inv};
  if (maskS > 0) {
    int s = (int)(row % maskS);
    #pragma unroll
    for (int i = 0; i < 4; ++i) if (lane * 4 + i > s) e[i] = -INFINITY;
  }
  float mx = fmaxf(fmaxf(e[0], e[1]), fmaxf(e[2], e[3]));
  #pragma unroll
  for (int o = 16; o; o >>= 1) mx = fmaxf(mx, __shfl_xor(mx, o, 32));
  float sum = 0.f;
  #pragma unroll
  for (int i = 0; i < 4; ++i) { e[i] = __expf(e[i] - mx); sum += e[i]; }
  #pragma unroll
  for (int o = 16; o; o >>= 1) sum += __shfl_xor(sum, o, 32);
  float rs = 1.0f / sum;
  ((float4*)p)[lane] = make_float4(e[0] * rs, e[1] * rs, e[2] * rs, e[3] * rs);
}

// ---------------- fused residual + LayerNorm over 1024-wide rows ----------------
__global__ __launch_bounds__(256) void add_ln_1024(const float* __restrict__ x,
                                                   const float* __restrict__ r,
                                                   const float* __restrict__ g,
                                                   const float* __restrict__ b,
                                                   float* __restrict__ out)
{
  __shared__ float s1[256], s2[256];
  ll row = blockIdx.x;
  int t = threadIdx.x;
  float4 xv = ((const float4*)(x + row * 1024))[t];
  float4 rv = ((const float4*)(r + row * 1024))[t];
  float4 v = make_float4(xv.x + rv.x, xv.y + rv.y, xv.z + rv.z, xv.w + rv.w);
  s1[t] = v.x + v.y + v.z + v.w;
  s2[t] = v.x * v.x + v.y * v.y + v.z * v.z + v.w * v.w;
  __syncthreads();
  for (int o = 128; o; o >>= 1) {
    if (t < o) { s1[t] += s1[t + o]; s2[t] += s2[t + o]; }
    __syncthreads();
  }
  float mean = s1[0] * (1.0f / 1024.0f);
  float var  = s2[0] * (1.0f / 1024.0f) - mean * mean;
  float rstd = rsqrtf(var + 1e-5f);
  float4 gv = ((const float4*)g)[t];
  float4 bv = ((const float4*)b)[t];
  float4 o4 = make_float4((v.x - mean) * rstd * gv.x + bv.x,
                          (v.y - mean) * rstd * gv.y + bv.y,
                          (v.z - mean) * rstd * gv.z + bv.z,
                          (v.w - mean) * rstd * gv.w + bv.w);
  ((float4*)(out + row * 1024))[t] = o4;
}

// ---------------- swish ----------------
__global__ void swish_k(float* __restrict__ x, ll n) {
  ll i = (ll)blockIdx.x * blockDim.x + threadIdx.x;
  ll stride = (ll)gridDim.x * blockDim.x;
  for (; i < n; i += stride) { float v = x[i]; x[i] = v / (1.0f + __expf(-v)); }
}

// ---------------- host orchestration ----------------
extern "C" void kernel_launch(void* const* d_in, const int* in_sizes, int n_in,
                              void* d_out, int out_size, void* d_ws, size_t ws_size,
                              hipStream_t stream) {
  const int Bb = 16, H = 8, D = 1024, DQ = 128, ST = 1024, SM = 2048, BN = 512, HID = 1024;
  const float INV_SCALE = 0.2973017787506803f;   // 1 / 128^(1/4)

  const float* mem = (const float*)d_in[0];
  const float* y   = (const float*)d_in[1];
  const float* Wq1 = (const float*)d_in[2];
  const float* Wk1 = (const float*)d_in[3];
  const float* Wv1 = (const float*)d_in[4];
  const float* Wo1 = (const float*)d_in[5];
  const float* Wq2 = (const float*)d_in[6];
  const float* Wk2 = (const float*)d_in[7];
  const float* Wv2 = (const float*)d_in[8];
  const float* Wo2 = (const float*)d_in[9];
  const float* E1  = (const float*)d_in[10];
  const float* D1  = (const float*)d_in[11];
  const float* E2  = (const float*)d_in[12];
  const float* D2  = (const float*)d_in[13];
  const float* g1  = (const float*)d_in[14];
  const float* b1  = (const float*)d_in[15];
  const float* g2  = (const float*)d_in[16];
  const float* b2  = (const float*)d_in[17];
  const float* g3  = (const float*)d_in[18];
  const float* b3  = (const float*)d_in[19];

  float* ws = (float*)d_ws;
  size_t off = 0;
  auto take = [&](size_t n) { float* p = ws + off; off += n; return p; };
  float* sK  = take((size_t)Bb * H * SM * DQ);   // 33.5M (mem-sized K slot)
  float* sV  = take((size_t)Bb * H * SM * DQ);   // 33.5M
  float* sQ  = take((size_t)Bb * H * ST * DQ);   // 16.8M
  float* sBm = take((size_t)Bb * H * ST * DQ);   // 16.8M
  float* sAt = take((size_t)Bb * ST * D);        // 16.8M (attn out)
  float* sY  = take((size_t)Bb * ST * D);        // 16.8M (yA)
  float* sAm = take((size_t)Bb * H * DQ * DQ);   // 2.1M  (A matrices)

  // host-side dispatch on compile-time loader specialization
  auto gemm = [&](const float* Ap, const float* Bp, float* Cp, int M, int N, int K,
                  ll ars, ll acs, ll aos, ll ais,
                  ll brs, ll bcs, ll bos, ll bis,
                  ll crs, ll cos_, ll cis, int Z, int idiv) {
    dim3 g((unsigned)(N / 128), (unsigned)(M / 128), (unsigned)Z);
    dim3 blk(256);
    bool ak = (acs == 1), bk = (brs == 1);
    if (ak && bk)
      gemm_bf16_wmma<true, true><<<g, blk, 0, stream>>>(Ap, Bp, Cp, M, N, K,
          ars, acs, aos, ais, brs, bcs, bos, bis, crs, cos_, cis, idiv);
    else if (ak && !bk)
      gemm_bf16_wmma<true, false><<<g, blk, 0, stream>>>(Ap, Bp, Cp, M, N, K,
          ars, acs, aos, ais, brs, bcs, bos, bis, crs, cos_, cis, idiv);
    else if (!ak && bk)
      gemm_bf16_wmma<false, true><<<g, blk, 0, stream>>>(Ap, Bp, Cp, M, N, K,
          ars, acs, aos, ais, brs, bcs, bos, bis, crs, cos_, cis, idiv);
    else
      gemm_bf16_wmma<false, false><<<g, blk, 0, stream>>>(Ap, Bp, Cp, M, N, K,
          ars, acs, aos, ais, brs, bcs, bos, bis, crs, cos_, cis, idiv);
  };
  auto softmax = [&](float* p, ll rows, int maskS) {
    softmax128_k<<<dim3((unsigned)((rows + 7) / 8)), dim3(256), 0, stream>>>(p, rows, maskS, INV_SCALE);
  };
  auto addln = [&](const float* x, const float* r, const float* g, const float* b, float* o, int rows) {
    add_ln_1024<<<dim3((unsigned)rows), dim3(256), 0, stream>>>(x, r, g, b, o);
  };

  // ================= self attention (masked) =================
  // projections: [b,h]: y[b] (1024x1024) x Wq1[h] (1024x128) -> [B,H,S,DQ]
  gemm(y, Wq1, sQ, ST, DQ, D,  D, 1, (ll)ST * D, 0,   DQ, 1, 0, (ll)D * DQ,
       DQ, (ll)H * ST * DQ, (ll)ST * DQ,  Bb * H, H);
  gemm(y, Wk1, sK, ST, DQ, D,  D, 1, (ll)ST * D, 0,   DQ, 1, 0, (ll)D * DQ,
       DQ, (ll)H * ST * DQ, (ll)ST * DQ,  Bb * H, H);
  gemm(y, Wv1, sV, ST, DQ, D,  D, 1, (ll)ST * D, 0,   DQ, 1, 0, (ll)D * DQ,
       DQ, (ll)H * ST * DQ, (ll)ST * DQ,  Bb * H, H);
  softmax(sK, (ll)Bb * H * ST, 0);
  softmax(sQ, (ll)Bb * H * ST, ST);      // Q gets the triangular head-dim mask
  // A = softK^T x V : per z=(b,h) 128x128, K=ST
  gemm(sK, sV, sAm, DQ, DQ, ST,  1, DQ, (ll)ST * DQ, 0,   DQ, 1, (ll)ST * DQ, 0,
       DQ, (ll)DQ * DQ, 0,  Bb * H, 1);
  // Bm = softQ x A
  gemm(sQ, sAm, sBm, ST, DQ, DQ,  DQ, 1, (ll)ST * DQ, 0,   DQ, 1, (ll)DQ * DQ, 0,
       DQ, (ll)ST * DQ, 0,  Bb * H, 1);
  // attn = view(Bm)[1024,1024] @ Wo1^T   (raw reshape == per-batch row-major view)
  gemm(sBm, Wo1, sAt, ST, D, D,  D, 1, (ll)ST * D, 0,   1, D, 0, 0,
       D, (ll)ST * D, 0,  Bb, 1);
  addln(y, sAt, g1, b1, sY, Bb * ST);    // yA

  // ================= cross attention =================
  gemm(sY,  Wq2, sQ, ST, DQ, D,  D, 1, (ll)ST * D, 0,   DQ, 1, 0, (ll)D * DQ,
       DQ, (ll)H * ST * DQ, (ll)ST * DQ,  Bb * H, H);
  gemm(mem, Wk2, sK, SM, DQ, D,  D, 1, (ll)SM * D, 0,   DQ, 1, 0, (ll)D * DQ,
       DQ, (ll)H * SM * DQ, (ll)SM * DQ,  Bb * H, H);
  gemm(mem, Wv2, sV, SM, DQ, D,  D, 1, (ll)SM * D, 0,   DQ, 1, 0, (ll)D * DQ,
       DQ, (ll)H * SM * DQ, (ll)SM * DQ,  Bb * H, H);
  softmax(sK, (ll)Bb * H * SM, 0);
  softmax(sQ, (ll)Bb * H * ST, 0);
  gemm(sK, sV, sAm, DQ, DQ, SM,  1, DQ, (ll)SM * DQ, 0,   DQ, 1, (ll)SM * DQ, 0,
       DQ, (ll)DQ * DQ, 0,  Bb * H, 1);
  gemm(sQ, sAm, sBm, ST, DQ, DQ,  DQ, 1, (ll)ST * DQ, 0,   DQ, 1, (ll)DQ * DQ, 0,
       DQ, (ll)ST * DQ, 0,  Bb * H, 1);
  gemm(sBm, Wo2, sAt, ST, D, D,  D, 1, (ll)ST * D, 0,   1, D, 0, 0,
       D, (ll)ST * D, 0,  Bb, 1);
  addln(sY, sAt, g2, b2, sQ, Bb * ST);   // yB -> reuse sQ slot

  // ================= LFFN: E1 -> D1 -> swish -> E2 -> D2 =================
  const float* yB = sQ;
  gemm(yB, E1, sBm, Bb * ST, BN, D,   D, 1, 0, 0,   1, D, 0, 0,   BN, 0, 0, 1, 1);
  gemm(sBm, D1, sAt, Bb * ST, HID, BN,  BN, 1, 0, 0,  1, BN, 0, 0,  HID, 0, 0, 1, 1);
  swish_k<<<dim3(2048), dim3(256), 0, stream>>>(sAt, (ll)Bb * ST * HID);
  gemm(sAt, E2, sBm, Bb * ST, BN, HID,  HID, 1, 0, 0,  1, HID, 0, 0,  BN, 0, 0, 1, 1);
  gemm(sBm, D2, sK, Bb * ST, D, BN,   BN, 1, 0, 0,   1, BN, 0, 0,   D, 0, 0, 1, 1);
  addln(yB, sK, g3, b3, (float*)d_out, Bb * ST);
}
